// Net_18391049961914
// MI455X (gfx1250) — matrix operand order repaired and checked by reading, not compiled, not run on hardware
//
#include <hip/hip_runtime.h>

typedef __attribute__((ext_vector_type(16))) __bf16 v16bf;
typedef __attribute__((ext_vector_type(8)))  __bf16 v8bf;
typedef __attribute__((ext_vector_type(8)))  float  v8f;
typedef unsigned short u16;
typedef unsigned int   u32;

__device__ __forceinline__ u16 f2bf(float f) {
    u32 u = __builtin_bit_cast(u32, f);
    u32 r = u + 0x7FFFu + ((u >> 16) & 1u);   // round-to-nearest-even
    return (u16)(r >> 16);
}
__device__ __forceinline__ float bf2f(u16 h) {
    u32 u = ((u32)h) << 16;
    return __builtin_bit_cast(float, u);
}
__device__ __forceinline__ __bf16 bfbits(u16 h) {
    return __builtin_bit_cast(__bf16, h);
}

// ---------------------------------------------------------------- cast kernels
__global__ void cast_f32_bf16(const float* __restrict__ in, u16* __restrict__ out, int n) {
    int i = blockIdx.x * blockDim.x + threadIdx.x;
    if (i < n) out[i] = f2bf(in[i]);
}

// One-time weight pack: f32 (CO,K) -> bf16 (CO_pad, K_pad), zero padded.
__global__ void pack_weights_bf16(const float* __restrict__ W, u16* __restrict__ out,
                                  int CO, int K, int CO_pad, int K_pad) {
    int i = blockIdx.x * blockDim.x + threadIdx.x;
    int total = CO_pad * K_pad;
    if (i >= total) return;
    int co = i / K_pad, k = i - co * K_pad;
    float v = (co < CO && k < K) ? W[(size_t)co * K + k] : 0.0f;
    out[i] = f2bf(v);
}

// ---- B-fragment load + WMMA step (bf16 padded weights, stride Kp) ----------
// B fragment (ISA 16-bit B layout): lane l -> col n = l&15, k = 8*(l>>4) + {0..7, 16..23}
template<int NT>
__device__ __forceinline__ void wmma_bstep(const v16bf& a, v8f (&acc)[NT],
                                           const u16* __restrict__ Wp, size_t Kp,
                                           int co_base, int l16, int hl, int k0)
{
#pragma unroll
    for (int t = 0; t < NT; ++t) {
        const u16* wrow = Wp + (size_t)(co_base + t * 16 + l16) * Kp + k0 + hl * 8;
        v8bf lo = *(const v8bf*)wrow;        // k .. k+7
        v8bf hi = *(const v8bf*)(wrow + 16); // k+16 .. k+23
        v16bf b = __builtin_shufflevector(lo, hi,
                      0, 1, 2, 3, 4, 5, 6, 7, 8, 9, 10, 11, 12, 13, 14, 15);
        acc[t] = __builtin_amdgcn_wmma_f32_16x16x32_bf16(
                     false, a, false, b, (short)0, acc[t], false, false);
    }
}

// ---- implicit-GEMM A-fragment gather (ISA 16-bit A layout) -----------------
// Closed-form k -> input offset; pure function of compile-time constants.
__device__ __forceinline__ constexpr int koff(int k, int KK, int KW, int IW, int IHIW) {
    int ci = k / KK;
    int r  = k - ci * KK;
    int kh = r / KW;
    int kw = r - kh * KW;
    return ci * IHIW + kh * IW + kw;
}

// One lane-half: hbase is a literal 0 or 8 at each call site, k0 is an
// unroll constant -> every offset folds to an immediate (zero address VALU).
template<bool GUARD>
__device__ __forceinline__ v16bf gather_half(const u16* __restrict__ abase,
                                             int k0, int hbase,
                                             int K, int KK, int KW, int IW, int IHIW)
{
    v16bf a;
#pragma unroll
    for (int e = 0; e < 16; ++e) {
        int k   = k0 + hbase + ((e < 8) ? e : e + 8);   // {0..7, 16..23}
        int off = koff(k, KK, KW, IW, IHIW);
        u16 v = (!GUARD || (k < K)) ? abase[off] : (u16)0;
        a[e] = bfbits(v);
    }
    return a;
}

// lane l -> row m = l&15, k = 8*(l>>4) + {0..7, 16..23}.
// Branch once on the lane-half; arms reconverge before the WMMA call.
template<bool GUARD, int KH, int KW, int IW, int IHIW>
__device__ __forceinline__ v16bf conv_gatherA(const u16* __restrict__ abase,
                                              int k0, int hl, int K)
{
    constexpr int KK = KH * KW;
    v16bf a;
    if (hl) a = gather_half<GUARD>(abase, k0, 8, K, KK, KW, IW, IHIW);
    else    a = gather_half<GUARD>(abase, k0, 0, K, KK, KW, IW, IHIW);
    return a;
}

// ------------------------------------------------- implicit-GEMM conv + BN/ReLU
// Fully static shape; unrolled k-chunk loop; immediate-offset A gather.
template<int NT, int CI, int IH, int IW, int CO, int KH, int KW, int OH, int OW>
__global__ __launch_bounds__(128) void conv_bn_relu_wmma(
    const u16*  __restrict__ in,  const u16* __restrict__ Wp,
    const float* __restrict__ cb, const float* __restrict__ bs,
    const float* __restrict__ bb, const float* __restrict__ bm,
    const float* __restrict__ bv, u16* __restrict__ out)
{
    constexpr int KK    = KH * KW;
    constexpr int K     = CI * KK;
    constexpr int Kp    = (K + 31) & ~31;
    constexpr int IHIW  = IH * IW;
    constexpr int OHOW  = OH * OW;
    constexpr int Kfull = K & ~31;

    const int tid  = threadIdx.x;
    const int wave = tid >> 5;
    const int lane = tid & 31;
    const int hl   = lane >> 4;
    const int l16  = lane & 15;

    const int pix_base = blockIdx.x * 64 + wave * 16;
    const int co_base  = blockIdx.y * (16 * NT);

    // pixel for this lane's A-fragment rows (grid is exact: always in range)
    const int pA   = pix_base + l16;
    int an   = pA / OHOW;
    int arem = pA - an * OHOW;
    int aoh  = arem / OW;
    int aow  = arem - aoh * OW;
    const u16* abase = in + (size_t)an * CI * IHIW + (size_t)aoh * IW + aow;

    v8f acc[NT];
#pragma unroll
    for (int t = 0; t < NT; ++t) acc[t] = (v8f){};

#pragma unroll
    for (int k0 = 0; k0 < Kfull; k0 += 32) {
        v16bf a = conv_gatherA<false, KH, KW, IW, IHIW>(abase, k0, hl, K);
        wmma_bstep<NT>(a, acc, Wp, Kp, co_base, l16, hl, k0);
        if (k0 + 32 < Kp)   // gfx1250 global_prefetch_b8 of next weight chunk
            __builtin_prefetch(Wp + (size_t)(co_base + l16) * Kp + k0 + 32, 0, 1);
    }
    if constexpr (K > Kfull) {   // guarded tail chunk (weights zero-padded)
        v16bf a = conv_gatherA<true, KH, KW, IW, IHIW>(abase, Kfull, hl, K);
        wmma_bstep<NT>(a, acc, Wp, Kp, co_base, l16, hl, Kfull);
    }

    // ---- epilogue: bias -> relu -> batchnorm, bf16 NCHW ----
#pragma unroll
    for (int t = 0; t < NT; ++t) {
        const int cog = co_base + t * 16 + l16;
        float scale = bs[cog] * rsqrtf(bv[cog] + 1e-5f);
        float bias  = cb[cog];
        float mean  = bm[cog];
        float beta  = bb[cog];
        int p0  = pix_base + hl * 8;   // C/D layout: row = vgpr + 8*half
        int n   = p0 / OHOW;
        int rem = p0 - n * OHOW;
#pragma unroll
        for (int r = 0; r < 8; ++r) {
            float v = acc[t][r] + bias;
            v = fmaxf(v, 0.0f);
            v = (v - mean) * scale + beta;
            out[((size_t)n * CO + cog) * OHOW + rem] = f2bf(v);
            if (++rem == OHOW) { rem = 0; ++n; }
        }
    }
}

// ---------------------------------------------------------------- 2x2 max pool
template<int C, int H, int W>
__global__ void pool2x2(const u16* __restrict__ in, u16* __restrict__ out, int total)
{
    int i = blockIdx.x * blockDim.x + threadIdx.x;
    if (i >= total) return;
    constexpr int OW2 = W >> 1, OH2 = H >> 1;
    int ow = i % OW2; int t = i / OW2;
    int oh = t % OH2; t /= OH2;
    int c  = t % C;   int n = t / C;
    const u16* base = in + ((size_t)(n * C + c) * H + oh * 2) * W + ow * 2;
    float a = bf2f(base[0]), b = bf2f(base[1]);
    float c2 = bf2f(base[W]), d = bf2f(base[W + 1]);
    out[i] = f2bf(fmaxf(fmaxf(a, b), fmaxf(c2, d)));
}

// ------------------------------------------------------------- FC GEMM (WMMA)
// A: M x K bf16 row-major (K % 32 == 0). Wp: bf16 (N_pad, K) zero padded.
// mode 0: ReLU -> bf16 out ; mode 1: linear -> f32 out
template<int NT, int N, int K>
__global__ __launch_bounds__(128) void fc_wmma(
    const u16* __restrict__ A, const u16* __restrict__ Wp,
    const float* __restrict__ bias, void* __restrict__ outp,
    int M, int mode)
{
    const int tid  = threadIdx.x;
    const int wave = tid >> 5;
    const int lane = tid & 31;
    const int hl   = lane >> 4;
    const int l16  = lane & 15;

    const int row_base = blockIdx.x * 64 + wave * 16;
    const int col_base = blockIdx.y * (16 * NT);

    const u16* arow = A + (size_t)(row_base + l16) * K;   // grid exact in M

    v8f acc[NT];
#pragma unroll
    for (int t = 0; t < NT; ++t) acc[t] = (v8f){};

    for (int c = 0; c < (K >> 5); ++c) {
        const int k0 = c << 5;

        const u16* ap = arow + k0 + hl * 8;
        v8bf lo = *(const v8bf*)ap;          // k .. k+7
        v8bf hi = *(const v8bf*)(ap + 16);   // k+16 .. k+23
        v16bf a = __builtin_shufflevector(lo, hi,
                      0, 1, 2, 3, 4, 5, 6, 7, 8, 9, 10, 11, 12, 13, 14, 15);

        wmma_bstep<NT>(a, acc, Wp, (size_t)K, col_base, l16, hl, k0);

        if (k0 + 32 < K)
            __builtin_prefetch(Wp + (size_t)(col_base + l16) * K + k0 + 32, 0, 1);
    }

#pragma unroll
    for (int t = 0; t < NT; ++t) {
        const int cog = col_base + t * 16 + l16;
        if (cog < N) {
            float bv = bias[cog];
#pragma unroll
            for (int r = 0; r < 8; ++r) {
                int p = row_base + hl * 8 + r;
                float v = acc[t][r] + bv;
                if (mode == 0) {
                    ((u16*)outp)[(size_t)p * N + cog] = f2bf(fmaxf(v, 0.0f));
                } else {
                    ((float*)outp)[(size_t)p * N + cog] = v;
                }
            }
        }
    }
}

// -------------------------------------- sigmoid + normalize + multiply (37 col)
__global__ void head_kernel(const float* __restrict__ logits,
                            float* __restrict__ out, int M)
{
    int r = blockIdx.x * blockDim.x + threadIdx.x;
    if (r >= M) return;
    float x[37];
#pragma unroll
    for (int j = 0; j < 37; ++j)
        x[j] = 1.0f / (1.0f + expf(-logits[r * 37 + j]));

    const int QS[11] = {0, 3, 5, 7, 9, 13, 15, 18, 25, 28, 31};
    const int QE[11] = {3, 5, 7, 9, 13, 15, 18, 25, 28, 31, 37};
    const int DP[11] = {-1, 1, 4, 4, 4, -1, 0, 13, 3, 7, 7};

    // _normalize: exact sequential semantics (seg_sum recomputed each j)
    for (int i = 0; i < 11; ++i) {
        int s = QS[i], e = QE[i];
        for (int j = s; j < e; ++j) {
            float seg = 0.0f;
            for (int t = s; t < e; ++t) seg += x[t];
            x[j] = x[j] / (seg + 1e-12f);
        }
    }
    // _multiply_prob
    for (int i = 0; i < 11; ++i) {
        int d = DP[i];
        if (d >= 0) {
            float m = x[d];
            for (int j = QS[i]; j < QE[i]; ++j) x[j] *= m;
        }
    }
#pragma unroll
    for (int j = 0; j < 37; ++j) out[r * 37 + j] = x[j];
}

// ============================================================================
extern "C" void kernel_launch(void* const* d_in, const int* in_sizes, int n_in,
                              void* d_out, int out_size, void* d_ws, size_t ws_size,
                              hipStream_t stream)
{
    const float* x   = (const float*)d_in[0];
    const float* c1w = (const float*)d_in[1];  const float* c1b = (const float*)d_in[2];
    const float* b1s = (const float*)d_in[3];  const float* b1b = (const float*)d_in[4];
    const float* b1m = (const float*)d_in[5];  const float* b1v = (const float*)d_in[6];
    const float* c2w = (const float*)d_in[7];  const float* c2b = (const float*)d_in[8];
    const float* b2s = (const float*)d_in[9];  const float* b2b = (const float*)d_in[10];
    const float* b2m = (const float*)d_in[11]; const float* b2v = (const float*)d_in[12];
    const float* c3w = (const float*)d_in[13]; const float* c3b = (const float*)d_in[14];
    const float* b3s = (const float*)d_in[15]; const float* b3b = (const float*)d_in[16];
    const float* b3m = (const float*)d_in[17]; const float* b3v = (const float*)d_in[18];
    const float* c4w = (const float*)d_in[19]; const float* c4b = (const float*)d_in[20];
    const float* b4s = (const float*)d_in[21]; const float* b4b = (const float*)d_in[22];
    const float* b4m = (const float*)d_in[23]; const float* b4v = (const float*)d_in[24];
    const float* f1w = (const float*)d_in[25]; const float* f1b = (const float*)d_in[26];
    const float* f2w = (const float*)d_in[27]; const float* f2b = (const float*)d_in[28];
    const float* f3w = (const float*)d_in[29]; const float* f3b = (const float*)d_in[30];

    const int N = in_sizes[0] / (3 * 60 * 60);   // batch (1024)

    // workspace: ping-pong bf16 activation buffers + f32 logits + packed weights
    char* ws = (char*)d_ws;
    const size_t offB = 177209344;               // max tenant of bufA (h2 pre-pool)
    const size_t offL = offB + 102760448;        // max tenant of bufB (h1)
    const size_t offW = offL + (size_t)N * 37 * 4;
    u16*   bufA   = (u16*)ws;
    u16*   bufB   = (u16*)(ws + offB);
    float* logits = (float*)(ws + offL);
    u16*   wbase  = (u16*)(ws + offW);

    // packed weight sub-allocations (bf16 elements), rows padded to K_pad
    u16* wc1 = wbase;                 // 16  x 96
    u16* wc2 = wc1 + 16 * 96;         // 32  x 416
    u16* wc3 = wc2 + 32 * 416;        // 64  x 288
    u16* wc4 = wc3 + 64 * 288;        // 128 x 576
    u16* wf1 = wc4 + 128 * 576;       // 1024 x 3200
    u16* wf2 = wf1 + 1024 * 3200;     // 512  x 1024
    u16* wf3 = wf2 + 512 * 1024;      // 48   x 512 (cols padded 37->48)

    // ---- one-time packs (tiny) ----
    {
        auto pk = [&](const float* w, u16* o, int CO, int K, int COp, int Kp) {
            int tot = COp * Kp;
            pack_weights_bf16<<<(tot + 255) / 256, 256, 0, stream>>>(w, o, CO, K, COp, Kp);
        };
        pk(c1w, wc1, 16, 75, 16, 96);
        pk(c2w, wc2, 32, 400, 32, 416);
        pk(c3w, wc3, 64, 288, 64, 288);
        pk(c4w, wc4, 128, 576, 128, 576);
        pk(f1w, wf1, 1024, 3200, 1024, 3200);
        pk(f2w, wf2, 512, 1024, 512, 1024);
        pk(f3w, wf3, 37, 512, 48, 512);
    }

    // x -> bf16
    {
        int n = N * 3 * 60 * 60;
        cast_f32_bf16<<<(n + 255) / 256, 256, 0, stream>>>(x, bufA, n);
    }
    // conv1: (N,3,60,60) -> (N,16,56,56)   CO=16 -> NT=1
    {
        int M = N * 56 * 56;
        conv_bn_relu_wmma<1, 3, 60, 60, 16, 5, 5, 56, 56>
            <<<dim3(M / 64, 1), 128, 0, stream>>>(
                bufA, wc1, c1b, b1s, b1b, b1m, b1v, bufB);
    }
    // conv2: (N,16,56,56) -> (N,32,52,52)  CO=32 -> NT=2
    {
        int M = N * 52 * 52;
        conv_bn_relu_wmma<2, 16, 56, 56, 32, 5, 5, 52, 52>
            <<<dim3(M / 64, 1), 128, 0, stream>>>(
                bufB, wc2, c2b, b2s, b2b, b2m, b2v, bufA);
    }
    // pool -> (N,32,26,26)
    {
        int total = N * 32 * 26 * 26;
        pool2x2<32, 52, 52><<<(total + 255) / 256, 256, 0, stream>>>(bufA, bufB, total);
    }
    // conv3: (N,32,26,26) -> (N,64,24,24)  CO=64 -> NT=4
    {
        int M = N * 24 * 24;
        conv_bn_relu_wmma<4, 32, 26, 26, 64, 3, 3, 24, 24>
            <<<dim3(M / 64, 1), 128, 0, stream>>>(
                bufB, wc3, c3b, b3s, b3b, b3m, b3v, bufA);
    }
    // pool -> (N,64,12,12)
    {
        int total = N * 64 * 12 * 12;
        pool2x2<64, 24, 24><<<(total + 255) / 256, 256, 0, stream>>>(bufA, bufB, total);
    }
    // conv4: (N,64,12,12) -> (N,128,10,10)  CO=128 -> NT=4, grid.y=2
    {
        int M = N * 10 * 10;
        conv_bn_relu_wmma<4, 64, 12, 12, 128, 3, 3, 10, 10>
            <<<dim3(M / 64, 2), 128, 0, stream>>>(
                bufB, wc4, c4b, b4s, b4b, b4m, b4v, bufA);
    }
    // pool -> (N,128,5,5) == flatten(N,3200)
    {
        int total = N * 128 * 5 * 5;
        pool2x2<128, 10, 10><<<(total + 255) / 256, 256, 0, stream>>>(bufA, bufB, total);
    }
    // fc1: 3200 -> 1024, relu   (NT=4, grid.y = 1024/64 = 16)
    fc_wmma<4, 1024, 3200><<<dim3(N / 64, 16), 128, 0, stream>>>(
        bufB, wf1, f1b, bufA, N, 0);
    // fc2: 1024 -> 512, relu    (NT=4, grid.y = 512/64 = 8)
    fc_wmma<4, 512, 1024><<<dim3(N / 64, 8), 128, 0, stream>>>(
        bufA, wf2, f2b, bufB, N, 0);
    // fc3: 512 -> 37 (cols padded to 48), linear f32 (NT=1, grid.y = 3)
    fc_wmma<1, 37, 512><<<dim3(N / 64, 3), 128, 0, stream>>>(
        bufB, wf3, f3b, logits, N, 1);
    // sigmoid + normalize + multiply_prob
    head_kernel<<<(N + 127) / 128, 128, 0, stream>>>(logits, (float*)d_out, N);
}
